// BertAttention_59639915872739
// MI455X (gfx1250) — matrix operand order repaired and checked by reading, not compiled, not run on hardware
//
#include <hip/hip_runtime.h>

// ---------------- constants ----------------
#define BATCH 8
#define SEQ   1024
#define DIM   768
#define HEADS 12
#define DHEAD 64
#define MROWS (BATCH*SEQ)          // 8192
#define NEG_INF (-1e30f)

typedef __attribute__((ext_vector_type(16))) __bf16 v16bf;
typedef __attribute__((ext_vector_type(8)))  __bf16 v8bf;
typedef __attribute__((ext_vector_type(8)))  float  v8f;
typedef __attribute__((ext_vector_type(4)))  float  v4f;
typedef __attribute__((ext_vector_type(4)))  int    v4i;

// ---------------- WMMA helpers ----------------
__device__ __forceinline__ v8f wmma_bf16(v16bf a, v16bf b, v8f c) {
  // D = A(16x32) * B(32x16) + C, f32 accumulate
  return __builtin_amdgcn_wmma_f32_16x16x32_bf16(false, a, false, b, (short)0, c,
                                                 false, false);
}

// A-fragment: 16x32 tile, rows = M, cols = K (row-major, lda in elements).
// lane<16 : row=lane,   K cols {0..7, 16..23}
// lane>=16: row=lane-16, K cols {8..15, 24..31}
__device__ __forceinline__ v16bf load_a_frag(const __bf16* base, int lda) {
  const int lane = threadIdx.x & 31;
  const int row  = lane & 15;
  const int cb   = (lane >> 4) * 8;
  const __bf16* p = base + (size_t)row * lda + cb;
  v8bf lo = *(const v8bf*)(p);
  v8bf hi = *(const v8bf*)(p + 16);
  v16bf a;
#pragma unroll
  for (int i = 0; i < 8; ++i) { a[i] = lo[i]; a[i + 8] = hi[i]; }
  return a;
}

// B-fragment: 32x16 tile, rows = K (reduction), cols = N (row-major, ldb elems).
// lane L holds column N=L%16, K rows {0..7,16..23} (L<16) or {8..15,24..31} (L>=16).
__device__ __forceinline__ v16bf load_b_frag(const __bf16* base, int ldb) {
  const int lane = threadIdx.x & 31;
  const int col  = lane & 15;
  const int cb   = (lane >> 4) * 8;
  v16bf b;
#pragma unroll
  for (int i = 0; i < 8; ++i) {
    b[i]     = base[(size_t)(cb + i) * ldb + col];
    b[i + 8] = base[(size_t)(cb + 16 + i) * ldb + col];
  }
  return b;
}

// ---------------- conversion kernels ----------------
__global__ __launch_bounds__(256) void cvt_x_kernel(const float* __restrict__ x,
                                                    __bf16* __restrict__ y, int n) {
  int i = blockIdx.x * 256 + threadIdx.x;
  if (i < n) y[i] = (__bf16)x[i];
}

// WT[k*768 + n] = W[n*768 + k]  (transpose so B-fragments are contiguous over n)
__global__ __launch_bounds__(256) void cvt_wt_kernel(const float* __restrict__ W,
                                                     __bf16* __restrict__ WT) {
  int i = blockIdx.x * 256 + threadIdx.x;
  if (i < DIM * DIM) {
    int k = i / DIM, n = i % DIM;
    WT[i] = (__bf16)W[n * DIM + k];
  }
}

// ---------------- fused QKV projection ----------------
// Each wave: one 16(M) x 64(N) output tile of one of {Q,K,V}.
// tiles per matrix = (8192/16)*(768/64) = 512*12 = 6144 ; 3 matrices ; 8 waves/block
__global__ __launch_bounds__(256) void qkv_kernel(
    const __bf16* __restrict__ X,
    const __bf16* __restrict__ WqT, const __bf16* __restrict__ WkT,
    const __bf16* __restrict__ WvT,
    const float* __restrict__ bq, const float* __restrict__ bk,
    const float* __restrict__ bv,
    __bf16* __restrict__ Q, __bf16* __restrict__ KT, __bf16* __restrict__ V) {
  const int waveg = blockIdx.x * 8 + (threadIdx.x >> 5);
  const int mat = waveg / 6144;
  const int t   = waveg % 6144;
  const int m0  = (t / 12) * 16;
  const int n0  = (t % 12) * 64;

  const __bf16* WT  = (mat == 0) ? WqT : (mat == 1) ? WkT : WvT;
  const float* bias = (mat == 0) ? bq  : (mat == 1) ? bk  : bv;

  v8f acc[4] = {};
  for (int k0 = 0; k0 < DIM; k0 += 32) {
    v16bf a = load_a_frag(X + (size_t)m0 * DIM + k0, DIM);
#pragma unroll
    for (int j = 0; j < 4; ++j) {
      v16bf b = load_b_frag(WT + (size_t)k0 * DIM + n0 + j * 16, DIM);
      acc[j] = wmma_bf16(a, b, acc[j]);
    }
  }

  const int lane  = threadIdx.x & 31;
  const int ncol  = lane & 15;
  const int rbase = (lane >> 4) * 8;
#pragma unroll
  for (int j = 0; j < 4; ++j) {
    const int n  = n0 + j * 16 + ncol;
    const float bb = bias[n];
    const int h  = n >> 6;       // head
    const int dh = n & 63;
#pragma unroll
    for (int r = 0; r < 8; ++r) {
      const int m = m0 + rbase + r;
      const int b_ = m >> 10, s = m & 1023;
      const __bf16 o = (__bf16)(acc[j][r] + bb);
      if (mat == 0)
        Q[((size_t)((b_ * HEADS + h) * SEQ + s) << 6) + dh] = o;         // [B,H,S,DH]
      else if (mat == 1)
        KT[((size_t)((b_ * HEADS + h) * DHEAD + dh) << 10) + s] = o;     // [B,H,DH,S]
      else
        V[((size_t)((b_ * HEADS + h) * SEQ + s) << 6) + dh] = o;         // [B,H,S,DH]
    }
  }
}

// ---------------- attention core ----------------
// One block (128 threads = 4 waves) per (b, h, 16-query tile).
// LDS: 16 x 1024 fp32 score strip (64 KB).
__global__ __launch_bounds__(128) void attn_kernel(
    const __bf16* __restrict__ Q, const __bf16* __restrict__ KT,
    const __bf16* __restrict__ V, const int* __restrict__ mask,
    float* __restrict__ probs, __bf16* __restrict__ Ctx) {
  __shared__ float sc[16 * SEQ];

  const int bid = blockIdx.x;          // b*H*64 + h*64 + qt
  const int qt = bid & 63;
  const int bh = bid >> 6;
  const int h = bh % HEADS, b = bh / HEADS;
  const int q0 = qt * 16;

  const int wave = threadIdx.x >> 5;
  const int lane = threadIdx.x & 31;
  const int ncol = lane & 15;
  const int rbase = (lane >> 4) * 8;

  const __bf16* qbase = Q + ((size_t)(b * HEADS + h) * SEQ + q0) * DHEAD;
  const __bf16* ktb   = KT + (size_t)(b * HEADS + h) * DHEAD * SEQ;
  const __bf16* vb    = V + (size_t)(b * HEADS + h) * SEQ * DHEAD;
  const int* mrow     = mask + b * SEQ;

  // --- phase 1: scores = Q K^T * scale (+mask) into LDS ---
  const v16bf a0 = load_a_frag(qbase, DHEAD);        // K-dim 0..31
  const v16bf a1 = load_a_frag(qbase + 32, DHEAD);   // K-dim 32..63
  const float scale = 0.125f;                        // 1/sqrt(64)

  for (int kt = wave; kt < 64; kt += 4) {
    const int key0 = kt * 16;
    v8f acc = {};
    acc = wmma_bf16(a0, load_b_frag(ktb + key0, SEQ), acc);
    acc = wmma_bf16(a1, load_b_frag(ktb + 32 * SEQ + key0, SEQ), acc);
    const int key = key0 + ncol;
    const bool keep = (mrow[key] != 0);
#pragma unroll
    for (int r = 0; r < 8; ++r)
      sc[(rbase + r) * SEQ + key] = keep ? acc[r] * scale : NEG_INF;
  }
  __syncthreads();

  // --- phase 2: softmax (8 threads per row, wave-shuffle reductions) ---
  const int row = threadIdx.x >> 3;   // 0..15
  const int sub = threadIdx.x & 7;
  float mx = -3.4e38f;
  for (int c = sub; c < SEQ; c += 8) mx = fmaxf(mx, sc[row * SEQ + c]);
#pragma unroll
  for (int o = 1; o < 8; o <<= 1) mx = fmaxf(mx, __shfl_xor(mx, o, 32));
  float sum = 0.f;
  for (int c = sub; c < SEQ; c += 8) {
    float e = __expf(sc[row * SEQ + c] - mx);
    sc[row * SEQ + c] = e;
    sum += e;
  }
#pragma unroll
  for (int o = 1; o < 8; o <<= 1) sum += __shfl_xor(sum, o, 32);
  const float inv = 1.f / sum;        // sum >= 1 always (exp(max-max)=1)
  __syncthreads();

  // --- phase 3: normalize, post-softmax mask, write probs ---
  // probs is 403 MB, written once and never re-read: use non-temporal stores
  // so it streams past WGP$/L2 instead of evicting resident Q/K/V/ctx tiles.
  float* pout = probs + ((size_t)(b * HEADS + h) * SEQ + q0) * SEQ + (size_t)row * SEQ;
  const int c0 = sub * 4;
#pragma unroll 4
  for (int j = 0; j < 32; ++j) {
    const int c = c0 + j * 32;
    v4f e4 = *(v4f*)&sc[row * SEQ + c];
    v4i m4 = *(const v4i*)&mrow[c];
    v4f o4;
    o4.x = m4.x ? e4.x * inv : 0.f;
    o4.y = m4.y ? e4.y * inv : 0.f;
    o4.z = m4.z ? e4.z * inv : 0.f;
    o4.w = m4.w ? e4.w * inv : 0.f;
    *(v4f*)&sc[row * SEQ + c] = o4;
    __builtin_nontemporal_store(o4, (v4f*)&pout[c]);
  }
  __syncthreads();

  // --- phase 4: ctx = probs @ V ; wave w covers dh range [w*16, w*16+16) ---
  const int n0 = wave * 16;
  v8f acc = {};
  for (int kc = 0; kc < 32; ++kc) {
    const int arow = lane & 15;
    const int cb = (lane >> 4) * 8 + kc * 32;
    v16bf a;
#pragma unroll
    for (int i = 0; i < 8; ++i) {
      a[i]     = (__bf16)sc[arow * SEQ + cb + i];
      a[i + 8] = (__bf16)sc[arow * SEQ + cb + 16 + i];
    }
    v16bf bf = load_b_frag(vb + (size_t)(kc * 32) * DHEAD + n0, DHEAD);
    acc = wmma_bf16(a, bf, acc);
  }
#pragma unroll
  for (int r = 0; r < 8; ++r) {
    const int q = q0 + rbase + r;
    Ctx[(size_t)(b * SEQ + q) * DIM + h * DHEAD + n0 + ncol] = (__bf16)acc[r];
  }
}

// ---------------- output projection + residual + LayerNorm ----------------
// One block (256 threads = 8 waves) per 16 rows; wave w covers 6 n-tiles.
__global__ __launch_bounds__(256) void out_kernel(
    const __bf16* __restrict__ Ctx, const __bf16* __restrict__ WoT,
    const float* __restrict__ bo, const float* __restrict__ hidden,
    const float* __restrict__ ln_g, const float* __restrict__ ln_b,
    float* __restrict__ out) {
  __shared__ float tile[16 * DIM];    // 48 KB

  const int m0 = blockIdx.x * 16;
  const int wave = threadIdx.x >> 5;
  const int lane = threadIdx.x & 31;
  const int ncol = lane & 15;
  const int rbase = (lane >> 4) * 8;

  v8f acc[6] = {};
  for (int k0 = 0; k0 < DIM; k0 += 32) {
    v16bf a = load_a_frag(Ctx + (size_t)m0 * DIM + k0, DIM);
#pragma unroll
    for (int nt = 0; nt < 6; ++nt) {
      const int n0 = wave * 96 + nt * 16;
      v16bf b = load_b_frag(WoT + (size_t)k0 * DIM + n0, DIM);
      acc[nt] = wmma_bf16(a, b, acc[nt]);
    }
  }
#pragma unroll
  for (int nt = 0; nt < 6; ++nt) {
    const int n = wave * 96 + nt * 16 + ncol;
    const float bias = bo[n];
#pragma unroll
    for (int r = 0; r < 8; ++r) {
      const int rl = rbase + r;
      tile[rl * DIM + n] = acc[nt][r] + bias + hidden[(size_t)(m0 + rl) * DIM + n];
    }
  }
  __syncthreads();

  // LayerNorm: 16 threads per row
  const int row = threadIdx.x >> 4;
  const int sub = threadIdx.x & 15;
  float s = 0.f;
  for (int c = sub; c < DIM; c += 16) s += tile[row * DIM + c];
#pragma unroll
  for (int o = 1; o < 16; o <<= 1) s += __shfl_xor(s, o, 32);
  const float mu = s * (1.f / DIM);
  float vs = 0.f;
  for (int c = sub; c < DIM; c += 16) {
    float d = tile[row * DIM + c] - mu;
    vs += d * d;
  }
#pragma unroll
  for (int o = 1; o < 16; o <<= 1) vs += __shfl_xor(vs, o, 32);
  const float inv = rsqrtf(vs * (1.f / DIM) + 1e-12f);
  const int m = m0 + row;
  for (int c = sub; c < DIM; c += 16) {
    float v = (tile[row * DIM + c] - mu) * inv;
    float o = v * ln_g[c] + ln_b[c];
    __builtin_nontemporal_store(o, &out[(size_t)m * DIM + c]);
  }
}

// ---------------- host launch ----------------
extern "C" void kernel_launch(void* const* d_in, const int* in_sizes, int n_in,
                              void* d_out, int out_size, void* d_ws, size_t ws_size,
                              hipStream_t stream) {
  const float* hidden = (const float*)d_in[0];
  const int*   mask   = (const int*)d_in[1];
  const float* Wq = (const float*)d_in[2];  const float* bq = (const float*)d_in[3];
  const float* Wk = (const float*)d_in[4];  const float* bk = (const float*)d_in[5];
  const float* Wv = (const float*)d_in[6];  const float* bv = (const float*)d_in[7];
  const float* Wo = (const float*)d_in[8];  const float* bo = (const float*)d_in[9];
  const float* lng = (const float*)d_in[10];
  const float* lnb = (const float*)d_in[11];

  float* out   = (float*)d_out;
  float* probs = out + (size_t)BATCH * SEQ * DIM;   // tuple: (out, probs)

  char* ws = (char*)d_ws;
  size_t off = 0;
  auto alloc = [&](size_t bytes) -> void* {
    void* p = ws + off;
    off = (off + bytes + 255) & ~(size_t)255;
    return p;
  };
  __bf16* Xbf  = (__bf16*)alloc((size_t)MROWS * DIM * 2);
  __bf16* WqT  = (__bf16*)alloc((size_t)DIM * DIM * 2);
  __bf16* WkT  = (__bf16*)alloc((size_t)DIM * DIM * 2);
  __bf16* WvT  = (__bf16*)alloc((size_t)DIM * DIM * 2);
  __bf16* WoT  = (__bf16*)alloc((size_t)DIM * DIM * 2);
  __bf16* Qb   = (__bf16*)alloc((size_t)MROWS * DIM * 2);   // [B,H,S,DH]
  __bf16* KTb  = (__bf16*)alloc((size_t)MROWS * DIM * 2);   // [B,H,DH,S]
  __bf16* Vb   = (__bf16*)alloc((size_t)MROWS * DIM * 2);   // [B,H,S,DH]
  __bf16* Ctxb = (__bf16*)alloc((size_t)MROWS * DIM * 2);   // [M, D]

  const int nX = MROWS * DIM;                 // 6291456
  cvt_x_kernel<<<(nX + 255) / 256, 256, 0, stream>>>(hidden, Xbf, nX);
  const int wblk = (DIM * DIM + 255) / 256;   // 2304
  cvt_wt_kernel<<<wblk, 256, 0, stream>>>(Wq, WqT);
  cvt_wt_kernel<<<wblk, 256, 0, stream>>>(Wk, WkT);
  cvt_wt_kernel<<<wblk, 256, 0, stream>>>(Wv, WvT);
  cvt_wt_kernel<<<wblk, 256, 0, stream>>>(Wo, WoT);

  // 3 * 6144 wave-tiles / 8 waves per block
  qkv_kernel<<<2304, 256, 0, stream>>>(Xbf, WqT, WkT, WvT, bq, bk, bv, Qb, KTb, Vb);

  // B*H*(S/16) = 6144 blocks
  attn_kernel<<<6144, 128, 0, stream>>>(Qb, KTb, Vb, mask, probs, Ctxb);

  // M/16 = 512 blocks
  out_kernel<<<512, 256, 0, stream>>>(Ctxb, WoT, bo, hidden, lng, lnb, out);
}
